// DGMoLEWrapper_79920751444278
// MI455X (gfx1250) — matrix operand order repaired and checked by reading, not compiled, not run on hardware
//
#include <hip/hip_runtime.h>

// MI455X / gfx1250, wave32. Main GEMM runs on v_wmma_f32_16x16x32_bf16 with an
// fp32 -> (bf16_hi, bf16_lo) split and 3-term accumulation for ~fp32 accuracy
// at 3/8 the cost of fp32 WMMA. Double-buffered LDS software pipeline:
// global loads for tile k+1 are in flight while tile k's 24 WMMAs execute.

typedef __attribute__((ext_vector_type(16))) __bf16 v16bf;
typedef __attribute__((ext_vector_type(8)))  __bf16 v8bf;
typedef __attribute__((ext_vector_type(8)))  float  v8f;
typedef __attribute__((ext_vector_type(4)))  float  v4f;

#define BM 128
#define BN 128
#define BK 32
#define LDSW 40   // padded LDS row stride (bf16 elems) -> conflict-free b128 reads

// ---------------------------------------------------------------------------
// Stage 1 of the pipeline: global fp32 -> registers. 256 threads fetch a
// 128x32 tile: thread t handles row t>>1, K-segment (t&1)*16 (16 floats).
// ---------------------------------------------------------------------------
__device__ __forceinline__ void load_regs(const float* __restrict__ src, int ld,
                                          int row0, int k0, int rows_valid,
                                          float v[16])
{
  const int t   = threadIdx.x;
  const int row = t >> 1;
  const int ks  = (t & 1) << 4;   // 0 or 16
  if (row < rows_valid) {
    const float* p = src + (size_t)(row0 + row) * (size_t)ld + (size_t)(k0 + ks);
    v4f a0 = *(const v4f*)(p + 0);
    v4f a1 = *(const v4f*)(p + 4);
    v4f a2 = *(const v4f*)(p + 8);
    v4f a3 = *(const v4f*)(p + 12);
    __builtin_prefetch(p + 2 * BK, 0, 0);   // hint tile k+2 -> global_prefetch
#pragma unroll
    for (int i = 0; i < 4; ++i) {
      v[i] = a0[i]; v[4 + i] = a1[i]; v[8 + i] = a2[i]; v[12 + i] = a3[i];
    }
  } else {
#pragma unroll
    for (int i = 0; i < 16; ++i) v[i] = 0.0f;
  }
}

// Stage 2: split fp32 regs into bf16 hi/lo planes and store to LDS.
__device__ __forceinline__ void store_lds(const float v[16],
                                          __bf16* dst_hi, __bf16* dst_lo)
{
  const int t   = threadIdx.x;
  const int row = t >> 1;
  const int ks  = (t & 1) << 4;
  v8bf hi0, hi1, lo0, lo1;
#pragma unroll
  for (int i = 0; i < 8; ++i) {
    __bf16 h0 = (__bf16)v[i];
    __bf16 h1 = (__bf16)v[8 + i];
    hi0[i] = h0; lo0[i] = (__bf16)(v[i]     - (float)h0);
    hi1[i] = h1; lo1[i] = (__bf16)(v[8 + i] - (float)h1);
  }
  const int off = row * LDSW + ks;
  *(v8bf*)(dst_hi + off)     = hi0;
  *(v8bf*)(dst_hi + off + 8) = hi1;
  *(v8bf*)(dst_lo + off)     = lo0;
  *(v8bf*)(dst_lo + off + 8) = lo1;
}

// A-operand fragment (16x32, M x K). ISA lane layout: lanes 0-15 (M=lane)
// hold K {0..7, 16..23}; lanes 16-31 (M=lane-16) hold K {8..15, 24..31}.
__device__ __forceinline__ v16bf frag_a(const __bf16* base, int rowbase)
{
  const int lane = threadIdx.x & 31;
  const int row  = rowbase + (lane & 15);
  const int koff = (lane >> 4) << 3;        // 0 or 8
  const __bf16* p = base + row * LDSW + koff;
  union { v16bf v; v8bf h[2]; } u;
  u.h[0] = *(const v8bf*)(p);               // K koff .. koff+7
  u.h[1] = *(const v8bf*)(p + 16);          // K koff+16 .. koff+23
  return u.v;
}

// B-operand fragment (32x16, K x N). Lanes 0-15 hold K {0..7, 16..23} of column lane;
// lanes 16-31 hold K {8..15, 24..31} of column lane-16. LDS tile stored [col][K].
__device__ __forceinline__ v16bf frag_b(const __bf16* base, int colbase)
{
  const int lane = threadIdx.x & 31;
  const int col  = colbase + (lane & 15);
  const int koff = (lane >> 4) << 3;        // 0 or 8
  const __bf16* p = base + col * LDSW + koff;
  union { v16bf v; v8bf h[2]; } u;
  u.h[0] = *(const v8bf*)(p);               // K koff .. koff+7
  u.h[1] = *(const v8bf*)(p + 16);          // K koff+16 .. koff+23
  return u.v;
}

// ---------------------------------------------------------------------------
// C[M,N] = A[M,K1] * B[N,K1]^T (+ A2[M,K2] * B2[N,K2]^T) (+ bias), fp32 in/out,
// computed via 3-term bf16 split WMMA. Block tile 128x128, 8 waves = 2(M)x4(N),
// each wave owns 64x32 = 4x2 16x16 accumulators. M must be a multiple of 128.
// Double-buffered LDS, one barrier per K-tile.
// ---------------------------------------------------------------------------
__global__ __launch_bounds__(256) void gemm3_kernel(
    const float* __restrict__ A,  int lda,
    const float* __restrict__ B,  int ldb,  int K1,
    const float* __restrict__ A2, int lda2,
    const float* __restrict__ B2, int ldb2, int K2,
    const float* __restrict__ bias,
    float* __restrict__ C, int ldc, int N)
{
  __shared__ __bf16 sAhi[2][BM * LDSW];
  __shared__ __bf16 sAlo[2][BM * LDSW];
  __shared__ __bf16 sBhi[2][BN * LDSW];
  __shared__ __bf16 sBlo[2][BN * LDSW];

  const int n0   = blockIdx.x * BN;
  const int m0   = blockIdx.y * BM;
  const int wid  = threadIdx.x >> 5;
  const int lane = threadIdx.x & 31;
  const int wm   = (wid >> 2) * 64;          // wave M offset: 0 / 64
  const int wn   = (wid & 3) * 32;           // wave N offset: 0/32/64/96
  const int bn_valid = (N - n0 < BN) ? (N - n0) : BN;

  const int t1     = K1 >> 5;                       // tiles in phase 1
  const int ntiles = t1 + ((A2 != nullptr) ? (K2 >> 5) : 0);

  v8f acc[4][2];
#pragma unroll
  for (int i = 0; i < 4; ++i)
#pragma unroll
    for (int j = 0; j < 2; ++j)
#pragma unroll
      for (int r = 0; r < 8; ++r) acc[i][j][r] = 0.0f;

  float va[16], vb[16];

  // Prologue: fetch tile 0 and stage it into buffer 0.
  load_regs(A, lda, m0, 0, BM, va);
  load_regs(B, ldb, n0, 0, bn_valid, vb);
  store_lds(va, sAhi[0], sAlo[0]);
  store_lds(vb, sBhi[0], sBlo[0]);

#pragma unroll 1
  for (int kt = 0; kt < ntiles; ++kt) {
    __syncthreads();   // buf[cur] staged by all waves; prior reads retired
    const int cur = kt & 1;
    const int nxt = cur ^ 1;

    // Issue global loads for tile kt+1 (complete while WMMAs run below).
    const bool have_next = (kt + 1) < ntiles;
    if (have_next) {
      const int  knext = kt + 1;
      const bool ph2   = knext >= t1;
      const float* pa  = ph2 ? A2 : A;
      const float* pb  = ph2 ? B2 : B;
      const int    la  = ph2 ? lda2 : lda;
      const int    lb  = ph2 ? ldb2 : ldb;
      const int    k0  = (ph2 ? (knext - t1) : knext) << 5;
      load_regs(pa, la, m0, k0, BM, va);
      load_regs(pb, lb, n0, k0, bn_valid, vb);
    }

    // Compute tile kt from buf[cur]: 24 WMMAs.
    {
      const __bf16* cAhi = sAhi[cur];
      const __bf16* cAlo = sAlo[cur];
      const __bf16* cBhi = sBhi[cur];
      const __bf16* cBlo = sBlo[cur];
      v16bf bh0 = frag_b(cBhi, wn + 0);
      v16bf bl0 = frag_b(cBlo, wn + 0);
      v16bf bh1 = frag_b(cBhi, wn + 16);
      v16bf bl1 = frag_b(cBlo, wn + 16);
#pragma unroll
      for (int i = 0; i < 4; ++i) {
        v16bf ah = frag_a(cAhi, wm + i * 16);
        v16bf al = frag_a(cAlo, wm + i * 16);
        acc[i][0] = __builtin_amdgcn_wmma_f32_16x16x32_bf16(
            false, ah, false, bh0, (short)0, acc[i][0], false, false);
        acc[i][0] = __builtin_amdgcn_wmma_f32_16x16x32_bf16(
            false, ah, false, bl0, (short)0, acc[i][0], false, false);
        acc[i][1] = __builtin_amdgcn_wmma_f32_16x16x32_bf16(
            false, ah, false, bh1, (short)0, acc[i][1], false, false);
        acc[i][1] = __builtin_amdgcn_wmma_f32_16x16x32_bf16(
            false, ah, false, bl1, (short)0, acc[i][1], false, false);
        acc[i][0] = __builtin_amdgcn_wmma_f32_16x16x32_bf16(
            false, al, false, bh0, (short)0, acc[i][0], false, false);
        acc[i][1] = __builtin_amdgcn_wmma_f32_16x16x32_bf16(
            false, al, false, bh1, (short)0, acc[i][1], false, false);
      }
    }

    // Stage tile kt+1 into buf[nxt] (safe: no one reads buf[nxt] until the
    // next barrier; our own buf[cur] fragment reads already completed).
    if (have_next) {
      store_lds(va, sAhi[nxt], sAlo[nxt]);
      store_lds(vb, sBhi[nxt], sBlo[nxt]);
    }
  }

  // Epilogue. C/D layout: VGPR r, lanes 0-15 -> M=r, N=lane; lanes 16-31 ->
  // M=r+8, N=lane-16.
  const int mhalf = (lane >> 4) << 3;        // 0 or 8
  const int nl    = lane & 15;
#pragma unroll
  for (int j = 0; j < 2; ++j) {
    const int n = n0 + wn + j * 16 + nl;
    if (n >= N) continue;
    const float bv = bias ? bias[n] : 0.0f;
#pragma unroll
    for (int i = 0; i < 4; ++i) {
      const int mbase = m0 + wm + i * 16 + mhalf;
#pragma unroll
      for (int r = 0; r < 8; ++r) {
        C[(size_t)(mbase + r) * (size_t)ldc + n] = acc[i][j][r] + bv;
      }
    }
  }
}

// ---------------------------------------------------------------------------
// Prep: Wcat[136][D]: rows 0-7 = W_router, rows 8+e*16+r = lora_A[e,:,r].
// ---------------------------------------------------------------------------
__global__ void prep_wcat_kernel(const float* __restrict__ Wr,
                                 const float* __restrict__ lA,
                                 float* __restrict__ Wcat, int D)
{
  const int idx = blockIdx.x * blockDim.x + threadIdx.x;
  const int total = 136 * D;
  if (idx >= total) return;
  const int rowi = idx / D;
  const int d    = idx - rowi * D;
  float v;
  if (rowi < 8) {
    v = Wr[(size_t)rowi * D + d];
  } else {
    const int er = rowi - 8;
    const int e  = er >> 4;
    const int r  = er & 15;
    v = lA[((size_t)e * D + d) * 16 + r];   // lora_A[e, d, r]
  }
  Wcat[idx] = v;
}

// Prep: LbT[o][er] = lora_B[er][o]  (lora_B flattened [128, Dout] row-major)
__global__ void prep_lbt_kernel(const float* __restrict__ lB,
                                float* __restrict__ LbT, int Dout)
{
  const int idx = blockIdx.x * blockDim.x + threadIdx.x;
  if (idx >= Dout * 128) return;
  const int o  = idx >> 7;
  const int er = idx & 127;
  LbT[idx] = lB[(size_t)er * Dout + o];
}

// ---------------------------------------------------------------------------
// Per-token sparsemax over 8 logits + scale LoRA activations:
// U[t, e*16+r] = w_e * h[t, e*16+r].  logitsH row: [8 logits | 128 h].
// ---------------------------------------------------------------------------
__global__ void router_sparsemax_kernel(const float* __restrict__ logitsH, int ldl,
                                        const float* __restrict__ b_router,
                                        float* __restrict__ U, int T)
{
  const int t = blockIdx.x * blockDim.x + threadIdx.x;
  if (t >= T) return;
  const float* row = logitsH + (size_t)t * ldl;
  float z[8], s[8];
#pragma unroll
  for (int e = 0; e < 8; ++e) { z[e] = row[e] + b_router[e]; s[e] = z[e]; }
  // insertion sort, descending
#pragma unroll
  for (int i = 1; i < 8; ++i) {
    float key = s[i];
    int j = i - 1;
    while (j >= 0 && s[j] < key) { s[j + 1] = s[j]; --j; }
    s[j + 1] = key;
  }
  float cum = 0.0f, tausum = 1.0f;
  int kz = 1;
#pragma unroll
  for (int k = 1; k <= 8; ++k) {
    cum += s[k - 1];
    if (1.0f + (float)k * s[k - 1] > cum) { kz = k; tausum = cum; }
  }
  const float tau = (tausum - 1.0f) / (float)kz;
  const float* h = row + 8;
  float* u = U + (size_t)t * 128;
#pragma unroll
  for (int e = 0; e < 8; ++e) {
    float w = z[e] - tau;
    w = (w > 0.0f) ? w : 0.0f;
    for (int r = 0; r < 16; ++r) u[e * 16 + r] = w * h[e * 16 + r];
  }
}

// ---------------------------------------------------------------------------
extern "C" void kernel_launch(void* const* d_in, const int* in_sizes, int n_in,
                              void* d_out, int out_size, void* d_ws, size_t ws_size,
                              hipStream_t stream)
{
  const float* x        = (const float*)d_in[0];
  const float* W_base   = (const float*)d_in[1];
  const float* b_base   = (const float*)d_in[2];
  const float* W_router = (const float*)d_in[3];
  const float* b_router = (const float*)d_in[4];
  const float* lora_A   = (const float*)d_in[5];
  const float* lora_B   = (const float*)d_in[6];
  float* out = (float*)d_out;

  const int D    = 4096;              // D_in  (in_sizes[3]/8)
  const int Dout = in_sizes[2];       // 4096
  const int T    = in_sizes[0] / D;   // 16384 tokens (multiple of 128)

  // Workspace layout (floats): ~22 MB total.
  float* ws      = (float*)d_ws;
  float* Wcat    = ws;                                  // 136 * D
  float* LbT     = Wcat + (size_t)136 * D;              // Dout * 128
  float* logitsH = LbT + (size_t)Dout * 128;            // T * 136
  float* U       = logitsH + (size_t)T * 136;           // T * 128
  (void)ws_size; (void)n_in; (void)out_size;

  // 1) Pack router weights + transposed lora_A into Wcat[136][D].
  {
    const int total = 136 * D;
    prep_wcat_kernel<<<(total + 255) / 256, 256, 0, stream>>>(W_router, lora_A, Wcat, D);
  }
  // 2) Transpose lora_B -> LbT[Dout][128].
  {
    const int total = Dout * 128;
    prep_lbt_kernel<<<(total + 255) / 256, 256, 0, stream>>>(lora_B, LbT, Dout);
  }
  // 3) logitsH[T,136] = x @ Wcat^T   (K = D)
  {
    dim3 grid((136 + BN - 1) / BN, T / BM);
    gemm3_kernel<<<grid, 256, 0, stream>>>(x, D, Wcat, D, D,
                                           nullptr, 0, nullptr, 0, 0,
                                           nullptr, logitsH, 136, 136);
  }
  // 4) Sparsemax + weight LoRA activations -> U[T,128].
  router_sparsemax_kernel<<<(T + 255) / 256, 256, 0, stream>>>(logitsH, 136, b_router, U, T);
  // 5) out = x @ W_base^T + U @ LbT^T + b_base.
  {
    dim3 grid((Dout + BN - 1) / BN, T / BM);
    gemm3_kernel<<<grid, 256, 0, stream>>>(x, D, W_base, D, D,
                                           U, 128, LbT, 128, 128,
                                           b_base, out, Dout, Dout);
  }
}
